// WorldModelMamba_27745488732935
// MI455X (gfx1250) — matrix operand order repaired and checked by reading, not compiled, not run on hardware
//
#include <hip/hip_runtime.h>
#include <hip/hip_bf16.h>
#include <math.h>

// ---------------------------------------------------------------------------
// WorldModelMamba forward for MI455X (gfx1250, wave32, WMMA).
// - Dense layers: v_wmma_f32_16x16x32_f16 GEMM, LDS-staged A, K-step 64,
//   2 N-tiles/wave, branch-free clamp+select padding.
// - Heavy convs: implicit-GEMM WMMA (im2col gather into LDS A-tile).
// - Heavy deconv: parity-split implicit-GEMM WMMA.
// ---------------------------------------------------------------------------

typedef __attribute__((ext_vector_type(16))) _Float16 v16h;
typedef __attribute__((ext_vector_type(8)))  float    v8f;

#define ACT_NONE     0
#define ACT_GELU     1
#define ACT_SOFTPLUS 2

__device__ __forceinline__ float gelu_f(float x) {
  float x3 = x * x * x;
  float t = tanhf(0.7978845608028654f * (x + 0.044715f * x3));
  return 0.5f * x * (1.0f + t);
}
__device__ __forceinline__ float silu_f(float x) {
  return x / (1.0f + expf(-x));
}
__device__ __forceinline__ float softplus_f(float x) {
  return (x > 20.0f) ? x : log1pf(expf(x));
}

__device__ __forceinline__ v8f wmma_f16(v16h a, v16h b, v8f c) {
  return __builtin_amdgcn_wmma_f32_16x16x32_f16(false, a, false, b, (short)0, c,
                                                false, false);
}

// A-fragment gather from LDS tile (contiguous k order: k = 16*hsel + e).
__device__ __forceinline__ void load_afrags(const _Float16* As, int mA, int hsel,
                                            v16h& a0, v16h& a1) {
#pragma unroll
  for (int e = 0; e < 16; ++e) {
    int kk = hsel * 16 + e;
    a0[e] = As[mA * 64 + kk];
    a1[e] = As[mA * 64 + 32 + kk];
  }
}

// B fragment: row n of Wt[N,K], 16 consecutive K values starting at k0.
template <bool KA, bool NA>
__device__ __forceinline__ v16h load_bfrag(const float* __restrict__ Wt,
                                           int K, int N, int n, int k0) {
  v16h b;
  const int nc = NA ? n : ((n < N) ? n : (N - 1));
  const float* wp = Wt + (size_t)nc * K;
#pragma unroll
  for (int e = 0; e < 16; ++e) {
    int gk = k0 + e;
    float v;
    if (KA) {
      v = wp[gk];
    } else {
      int gkc = (gk < K) ? gk : (K - 1);
      v = wp[gkc];
      v = (gk < K) ? v : 0.0f;
    }
    if (!NA) v = (n < N) ? v : 0.0f;
    b[e] = (_Float16)v;
  }
  return b;
}

// ---------------------------------------------------------------------------
// GEMM: Y[M,N] = act( X[M,K] @ W[N,K]^T + bias ).
// Block = 256 threads = 8 waves; block tile 16 x 256 (32 cols/wave, 2 acc).
// ---------------------------------------------------------------------------
template <bool KA, bool NA>
__global__ __launch_bounds__(256) void gemm_xwT_wmma(
    const float* __restrict__ X, int ldx,
    const float* __restrict__ Wt,    // [N,K] row-major
    const float* __restrict__ bias,  // may be null
    float* __restrict__ Y, int ldy,
    int M, int N, int K, int act) {
  __shared__ _Float16 As[16 * 64];
  const int tid  = threadIdx.x;
  const int lane = tid & 31;
  const int wave = tid >> 5;
  const int m0   = blockIdx.y * 16;
  const int n0   = blockIdx.x * 256 + wave * 32;
  const int mA   = lane & 15;
  const int hsel = lane >> 4;
  const int nB0  = n0 + (lane & 15);
  const int nB1  = nB0 + 16;

  const int ar   = (tid * 4) >> 6;   // A staging: 4 consecutive cols, one row
  const int ak   = (tid * 4) & 63;
  const int gmA  = m0 + ar;
  const int gmAc = (gmA < M) ? gmA : (M - 1);
  const bool mok = (gmA < M);
  const float* xrow = X + (size_t)gmAc * ldx;

  v8f acc0 = {}, acc1 = {};

  for (int kb = 0; kb < K; kb += 64) {
    __syncthreads();
#pragma unroll
    for (int j = 0; j < 4; ++j) {
      int gk = kb + ak + j;
      float v;
      if (KA) {
        v = xrow[gk];
      } else {
        int gkc = (gk < K) ? gk : (K - 1);
        v = xrow[gkc];
        v = (gk < K) ? v : 0.0f;
      }
      v = mok ? v : 0.0f;
      As[ar * 64 + ak + j] = (_Float16)v;
    }
    __syncthreads();

    v16h a0, a1;
    load_afrags(As, mA, hsel, a0, a1);

    const int kh0 = kb + hsel * 16;
    const int kh1 = kb + 32 + hsel * 16;
    v16h b00 = load_bfrag<KA, NA>(Wt, K, N, nB0, kh0);
    v16h b01 = load_bfrag<KA, NA>(Wt, K, N, nB0, kh1);
    v16h b10 = load_bfrag<KA, NA>(Wt, K, N, nB1, kh0);
    v16h b11 = load_bfrag<KA, NA>(Wt, K, N, nB1, kh1);

    if (kb + 64 < K) {
      const int npf = NA ? nB0 : ((nB0 < N) ? nB0 : (N - 1));
      __builtin_prefetch(Wt + (size_t)npf * K + kb + 64, 0, 1);
    }

    acc0 = wmma_f16(a0, b00, acc0);
    acc0 = wmma_f16(a1, b01, acc0);
    acc1 = wmma_f16(a0, b10, acc1);
    acc1 = wmma_f16(a1, b11, acc1);
  }

  float bv0 = 0.0f, bv1 = 0.0f;
  if (bias) {
    bv0 = (NA || nB0 < N) ? bias[NA ? nB0 : ((nB0 < N) ? nB0 : 0)] : 0.0f;
    bv1 = (NA || nB1 < N) ? bias[NA ? nB1 : ((nB1 < N) ? nB1 : 0)] : 0.0f;
  }
#pragma unroll
  for (int r = 0; r < 8; ++r) {
    int mOut = m0 + r + 8 * hsel;
    if (mOut >= M) continue;
    float v0 = acc0[r] + bv0;
    float v1 = acc1[r] + bv1;
    if (act == ACT_GELU)          { v0 = gelu_f(v0);     v1 = gelu_f(v1); }
    else if (act == ACT_SOFTPLUS) { v0 = softplus_f(v0); v1 = softplus_f(v1); }
    if (NA || nB0 < N) Y[(size_t)mOut * ldy + nB0] = v0;
    if (NA || nB1 < N) Y[(size_t)mOut * ldy + nB1] = v1;
  }
}

// ---------------------------------------------------------------------------
// Implicit-GEMM 3x3 conv (pad 1) + GELU, WMMA.
// M = NF*H*W output pixels, N = Cout, K = Cin*9.
// Weights (Cout,Cin,3,3) are already row-major [Cout][K] -> contiguous B loads.
// A-tile is an im2col gather into LDS (clamped, branch-free).
// Block tile: 16 pixels x 128 output channels (8 waves x 16).
// ---------------------------------------------------------------------------
template <bool KA>
__global__ __launch_bounds__(256) void conv3x3_igemm_wmma(
    const float* __restrict__ x,   // [NF,Cin,H,W]
    const float* __restrict__ w,   // [Cout,Cin,3,3]
    const float* __restrict__ bias,
    float* __restrict__ y,         // [NF,Cout,H,W]
    int NF, int Cin, int H, int W, int Cout) {
  const int K  = Cin * 9;
  const int HW = H * W;
  const int M  = NF * HW;
  __shared__ _Float16 As[16 * 64];
  const int tid  = threadIdx.x;
  const int lane = tid & 31;
  const int wave = tid >> 5;
  const int m0   = blockIdx.y * 16;
  const int co0  = blockIdx.x * 128 + wave * 16;
  const int mA   = lane & 15;
  const int hsel = lane >> 4;
  const int coB  = co0 + (lane & 15);
  const int coBc = (coB < Cout) ? coB : (Cout - 1);

  const int ar  = (tid * 4) >> 6;
  const int ak  = (tid * 4) & 63;
  const int gm  = m0 + ar;
  const int gmc = (gm < M) ? gm : (M - 1);
  const bool mok = (gm < M);
  const int n_img = gmc / HW;
  const int hwrem = gmc % HW;
  const int hA    = hwrem / W;
  const int wA    = hwrem % W;
  const float* xbase = x + (size_t)n_img * Cin * HW;

  v8f acc = {};
  for (int kb = 0; kb < K; kb += 64) {
    __syncthreads();
#pragma unroll
    for (int j = 0; j < 4; ++j) {
      int k  = kb + ak + j;
      int kc = KA ? k : ((k < K) ? k : (K - 1));
      int ci = kc / 9, t9 = kc % 9;
      int kh = t9 / 3, kw = t9 % 3;
      int ih = hA + kh - 1, iw = wA + kw - 1;
      bool ok = mok && ((unsigned)ih < (unsigned)H) && ((unsigned)iw < (unsigned)W) &&
                (KA || (k < K));
      int ihc = ((unsigned)ih < (unsigned)H) ? ih : 0;
      int iwc = ((unsigned)iw < (unsigned)W) ? iw : 0;
      float v = xbase[((size_t)ci * H + ihc) * W + iwc];
      As[ar * 64 + ak + j] = (_Float16)(ok ? v : 0.0f);
    }
    __syncthreads();

    v16h a0, a1;
    load_afrags(As, mA, hsel, a0, a1);

    const float* wp = w + (size_t)coBc * K;
    const int kh0 = kb + hsel * 16;
    const int kh1 = kb + 32 + hsel * 16;
    v16h b0, b1;
#pragma unroll
    for (int e = 0; e < 16; ++e) {
      int g0 = kh0 + e, g1 = kh1 + e;
      float v0, v1;
      if (KA) { v0 = wp[g0]; v1 = wp[g1]; }
      else {
        v0 = wp[(g0 < K) ? g0 : (K - 1)]; v0 = (g0 < K) ? v0 : 0.0f;
        v1 = wp[(g1 < K) ? g1 : (K - 1)]; v1 = (g1 < K) ? v1 : 0.0f;
      }
      b0[e] = (_Float16)v0;
      b1[e] = (_Float16)v1;
    }
    acc = wmma_f16(a0, b0, acc);
    acc = wmma_f16(a1, b1, acc);
  }

  if (coB >= Cout) return;
  float bv = bias[coB];
#pragma unroll
  for (int r = 0; r < 8; ++r) {
    int mOut = m0 + r + 8 * hsel;
    if (mOut >= M) continue;
    int n_o = mOut / HW, rem = mOut % HW;
    int h_o = rem / W, w_o = rem % W;
    y[((size_t)(n_o * Cout + coB) * H + h_o) * W + w_o] = gelu_f(acc[r] + bv);
  }
}

// ---------------------------------------------------------------------------
// Implicit-GEMM transposed conv 4x4 stride 2 pad 1 + GELU, parity-split.
// For output parity (ph,pw): oh = 2*mh+ph, ow = 2*mw+pw,
//   ih = mh + ph - jh, iw = mw + pw - jw, tap kh = (1-ph) + 2*jh (jh in {0,1}).
// M = NF*H*W (input-sized per parity class), K = Cin*4, N = Cout.
// B is a gather from w[ci][Cout][4][4].  Requires K % 64 == 0.
// ---------------------------------------------------------------------------
__global__ __launch_bounds__(256) void deconv_igemm_wmma(
    const float* __restrict__ x,   // [NF,Cin,H,W]
    const float* __restrict__ w,   // [Cin,Cout,4,4]
    const float* __restrict__ bias,
    float* __restrict__ y,         // [NF,Cout,2H,2W]
    int NF, int Cin, int H, int W, int Cout, int ph, int pw) {
  const int K  = Cin * 4;
  const int HW = H * W;
  const int M  = NF * HW;
  const int Ho = 2 * H, Wo = 2 * W;
  __shared__ _Float16 As[16 * 64];
  const int tid  = threadIdx.x;
  const int lane = tid & 31;
  const int wave = tid >> 5;
  const int m0   = blockIdx.y * 16;
  const int co0  = blockIdx.x * 128 + wave * 16;
  const int mA   = lane & 15;
  const int hsel = lane >> 4;
  const int coB  = co0 + (lane & 15);
  const int coBc = (coB < Cout) ? coB : (Cout - 1);

  const int ar  = (tid * 4) >> 6;
  const int ak  = (tid * 4) & 63;
  const int gm  = m0 + ar;
  const int gmc = (gm < M) ? gm : (M - 1);
  const bool mok = (gm < M);
  const int n_img = gmc / HW;
  const int hwrem = gmc % HW;
  const int mh    = hwrem / W;
  const int mw    = hwrem % W;
  const float* xbase = x + (size_t)n_img * Cin * HW;

  v8f acc = {};
  for (int kb = 0; kb < K; kb += 64) {
    __syncthreads();
#pragma unroll
    for (int j = 0; j < 4; ++j) {
      int k  = kb + ak + j;
      int ci = k >> 2, t4 = k & 3;
      int jh = t4 >> 1, jw = t4 & 1;
      int ih = mh + ph - jh, iw = mw + pw - jw;
      bool ok = mok && ((unsigned)ih < (unsigned)H) && ((unsigned)iw < (unsigned)W);
      int ihc = ((unsigned)ih < (unsigned)H) ? ih : 0;
      int iwc = ((unsigned)iw < (unsigned)W) ? iw : 0;
      float v = xbase[((size_t)ci * H + ihc) * W + iwc];
      As[ar * 64 + ak + j] = (_Float16)(ok ? v : 0.0f);
    }
    __syncthreads();

    v16h a0, a1;
    load_afrags(As, mA, hsel, a0, a1);

    // B gather: k -> (ci, jh, jw), tap (kh,kw) = ((1-ph)+2jh, (1-pw)+2jw)
    const int qh = 1 - ph, qw = 1 - pw;
    v16h b0, b1;
#pragma unroll
    for (int e = 0; e < 16; ++e) {
      int g0 = kb + hsel * 16 + e;
      int g1 = g0 + 32;
      int ci0 = g0 >> 2, t0 = g0 & 3;
      int ci1 = g1 >> 2, t1 = g1 & 3;
      int kh0 = qh + 2 * (t0 >> 1), kw0 = qw + 2 * (t0 & 1);
      int kh1 = qh + 2 * (t1 >> 1), kw1 = qw + 2 * (t1 & 1);
      b0[e] = (_Float16)w[((size_t)(ci0 * Cout + coBc) * 4 + kh0) * 4 + kw0];
      b1[e] = (_Float16)w[((size_t)(ci1 * Cout + coBc) * 4 + kh1) * 4 + kw1];
    }
    acc = wmma_f16(a0, b0, acc);
    acc = wmma_f16(a1, b1, acc);
  }

  if (coB >= Cout) return;
  float bv = bias[coB];
#pragma unroll
  for (int r = 0; r < 8; ++r) {
    int mOut = m0 + r + 8 * hsel;
    if (mOut >= M) continue;
    int n_o = mOut / HW, rem = mOut % HW;
    int h_o = 2 * (rem / W) + ph, w_o = 2 * (rem % W) + pw;
    y[((size_t)(n_o * Cout + coB) * Ho + h_o) * Wo + w_o] = gelu_f(acc[r] + bv);
  }
}

// ---------------------------------------------------------------------------
// Direct 3x3 conv + GELU (small-channel cases). One thread per output.
// ---------------------------------------------------------------------------
__global__ void conv3x3_gelu(const float* __restrict__ x, const float* __restrict__ w,
                             const float* __restrict__ b, float* __restrict__ y,
                             int N, int Cin, int H, int W, int Cout) {
  int idx = blockIdx.x * blockDim.x + threadIdx.x;
  int total = N * Cout * H * W;
  if (idx >= total) return;
  int wo = idx % W;  int t = idx / W;
  int ho = t % H;    t /= H;
  int co = t % Cout; int n = t / Cout;
  float acc = b[co];
  for (int ci = 0; ci < Cin; ++ci) {
    const float* xp = x + ((size_t)(n * Cin + ci) * H) * W;
    const float* wp = w + ((size_t)(co * Cin + ci) * 9);
#pragma unroll
    for (int kh = 0; kh < 3; ++kh) {
      int ih = ho + kh - 1;
      if ((unsigned)ih >= (unsigned)H) continue;
#pragma unroll
      for (int kw = 0; kw < 3; ++kw) {
        int iw = wo + kw - 1;
        if ((unsigned)iw >= (unsigned)W) continue;
        acc += xp[(size_t)ih * W + iw] * wp[kh * 3 + kw];
      }
    }
  }
  y[idx] = gelu_f(acc);
}

// 2x2 max pool, stride 2, VALID
__global__ void maxpool2_k(const float* __restrict__ x, float* __restrict__ y,
                           int NC, int H, int W) {
  int Ho = H / 2, Wo = W / 2;
  int idx = blockIdx.x * blockDim.x + threadIdx.x;
  int total = NC * Ho * Wo;
  if (idx >= total) return;
  int wo = idx % Wo; int t = idx / Wo;
  int ho = t % Ho;   int c = t / Ho;
  const float* xp = x + ((size_t)c * H + ho * 2) * W + wo * 2;
  y[idx] = fmaxf(fmaxf(xp[0], xp[1]), fmaxf(xp[W], xp[W + 1]));
}

// Direct transposed conv 4x4 s2 p1 + GELU, optional post-GELU residual.
__global__ void deconv4x4s2_gelu(const float* __restrict__ x, const float* __restrict__ w,
                                 const float* __restrict__ b, const float* __restrict__ res,
                                 float* __restrict__ y,
                                 int N, int Cin, int H, int W, int Cout) {
  int Ho = 2 * H, Wo = 2 * W;
  int idx = blockIdx.x * blockDim.x + threadIdx.x;
  int total = N * Cout * Ho * Wo;
  if (idx >= total) return;
  int ow = idx % Wo; int t = idx / Wo;
  int oh = t % Ho;   t /= Ho;
  int co = t % Cout; int n = t / Cout;
  float acc = b[co];
  for (int ci = 0; ci < Cin; ++ci) {
    const float* xp = x + ((size_t)(n * Cin + ci) * H) * W;
    const float* wp = w + ((size_t)(ci * Cout + co) * 16);
#pragma unroll
    for (int kh = 0; kh < 4; ++kh) {
      int th = oh + 1 - kh;
      if (th < 0 || (th & 1)) continue;
      int ih = th >> 1;
      if (ih >= H) continue;
#pragma unroll
      for (int kw = 0; kw < 4; ++kw) {
        int tw = ow + 1 - kw;
        if (tw < 0 || (tw & 1)) continue;
        int iw = tw >> 1;
        if (iw >= W) continue;
        acc += xp[(size_t)ih * W + iw] * wp[kh * 4 + kw];
      }
    }
  }
  float v = gelu_f(acc);
  if (res) v += res[idx];
  y[idx] = v;
}

// Mamba depthwise causal conv (window 4, left pad 3) + SiLU.
__global__ void mamba_conv_silu(const float* __restrict__ xz,
                                const float* __restrict__ cw,
                                const float* __restrict__ cb,
                                float* __restrict__ xi, int L, int D) {
  int idx = blockIdx.x * blockDim.x + threadIdx.x;
  if (idx >= L * D) return;
  int d = idx % D, l = idx / D;
  float acc = cb[d];
#pragma unroll
  for (int j = 0; j < 4; ++j) {
    int li = l - 3 + j;
    if (li < 0) continue;
    acc += xz[(size_t)li * 4096 + d] * cw[d * 4 + j];
  }
  xi[idx] = silu_f(acc);
}

// Selective scan: one thread per channel, 16-wide state in registers.
__global__ void mamba_scan(const float* __restrict__ delta,
                           const float* __restrict__ dbc,
                           const float* __restrict__ xi,
                           const float* __restrict__ xz,
                           const float* __restrict__ A_log,
                           const float* __restrict__ Dp,
                           float* __restrict__ g, int L, int D) {
  int d = blockIdx.x * blockDim.x + threadIdx.x;
  if (d >= D) return;
  float A[16], h[16];
#pragma unroll
  for (int s = 0; s < 16; ++s) { A[s] = -expf(A_log[d * 16 + s]); h[s] = 0.0f; }
  float Dv = Dp[d];
  for (int l = 0; l < L; ++l) {
    float dl = delta[(size_t)l * D + d];
    float xv = xi[(size_t)l * D + d];
    const float* Brow = dbc + (size_t)l * 96 + 64;
    const float* Crow = dbc + (size_t)l * 96 + 80;
    float yacc = 0.0f;
#pragma unroll
    for (int s = 0; s < 16; ++s) {
      float dA = expf(dl * A[s]);
      h[s] = dA * h[s] + dl * Brow[s] * xv;
      yacc += h[s] * Crow[s];
    }
    float yv = yacc + xv * Dv;
    float rv = xz[(size_t)l * 4096 + 2048 + d];
    g[(size_t)l * D + d] = yv * silu_f(rv);
  }
}

__global__ void copy_f32(const float* __restrict__ src, float* __restrict__ dst, int n) {
  int i = blockIdx.x * blockDim.x + threadIdx.x;
  if (i < n) dst[i] = src[i];
}

// ---------------------------------------------------------------------------
// Host orchestration
// ---------------------------------------------------------------------------
static inline int cdiv_i(int a, int b) { return (a + b - 1) / b; }

static inline void run_gemm(hipStream_t s, const float* X, int ldx, const float* W,
                            const float* bias, float* Y, int ldy,
                            int M, int N, int K, int act) {
  dim3 grid(cdiv_i(N, 256), cdiv_i(M, 16));
  dim3 blk(256);
  bool ka = (K % 64) == 0;
  bool na = (N % 256) == 0;
  if (ka && na)
    gemm_xwT_wmma<true, true><<<grid, blk, 0, s>>>(X, ldx, W, bias, Y, ldy, M, N, K, act);
  else if (ka)
    gemm_xwT_wmma<true, false><<<grid, blk, 0, s>>>(X, ldx, W, bias, Y, ldy, M, N, K, act);
  else if (na)
    gemm_xwT_wmma<false, true><<<grid, blk, 0, s>>>(X, ldx, W, bias, Y, ldy, M, N, K, act);
  else
    gemm_xwT_wmma<false, false><<<grid, blk, 0, s>>>(X, ldx, W, bias, Y, ldy, M, N, K, act);
}

static inline void run_conv_igemm(hipStream_t s, const float* x, const float* w,
                                  const float* b, float* y,
                                  int NF, int Cin, int H, int W, int Cout) {
  int M = NF * H * W, K = Cin * 9;
  dim3 grid(cdiv_i(Cout, 128), cdiv_i(M, 16));
  if ((K % 64) == 0)
    conv3x3_igemm_wmma<true><<<grid, dim3(256), 0, s>>>(x, w, b, y, NF, Cin, H, W, Cout);
  else
    conv3x3_igemm_wmma<false><<<grid, dim3(256), 0, s>>>(x, w, b, y, NF, Cin, H, W, Cout);
}

extern "C" void kernel_launch(void* const* d_in, const int* in_sizes, int n_in,
                              void* d_out, int out_size, void* d_ws, size_t ws_size,
                              hipStream_t stream) {
  (void)in_sizes; (void)n_in; (void)out_size; (void)ws_size;
  #define IN(i) ((const float*)d_in[i])

  const float* obs      = IN(0);
  const float* action   = IN(1);
  const float* state_t  = IN(2);
  const float* steps_l  = IN(3);
  const float* cur_r    = IN(4);
  const float* e0w0 = IN(5);  const float* e0b0 = IN(6);
  const float* e0w1 = IN(7);  const float* e0b1 = IN(8);
  const float* e1w0 = IN(9);  const float* e1b0 = IN(10);
  const float* e1w1 = IN(11); const float* e1b1 = IN(12);
  const float* e1w2 = IN(13); const float* e1b2 = IN(14);
  const float* e1w3 = IN(15); const float* e1b3 = IN(16);
  const float* lw0 = IN(17); const float* lb0 = IN(18);
  const float* lw1 = IN(19); const float* lb1 = IN(20);
  const float* lw2 = IN(21); const float* lb2 = IN(22);
  const float* lw3 = IN(23); const float* lb3 = IN(24);
  const float* dmw0 = IN(61); const float* dmb0 = IN(62);
  const float* dmw1 = IN(63); const float* dmb1 = IN(64);
  const float* c0w0 = IN(65); const float* c0b0 = IN(66);
  const float* c0w1 = IN(67); const float* c0b1 = IN(68);
  const float* c0w2 = IN(69); const float* c0b2 = IN(70);
  const float* c1w0 = IN(71); const float* c1b0 = IN(72);
  const float* c1w1 = IN(73); const float* c1b1 = IN(74);
  const float* c1w2 = IN(75); const float* c1b2 = IN(76);
  const float* d0w = IN(77); const float* d0b = IN(78);
  const float* d1w = IN(79); const float* d1b = IN(80);
  const float* d2w = IN(81); const float* d2b = IN(82);

  float* ws  = (float*)d_ws;
  float* out = (float*)d_out;

  const size_t BUF1  = 0;           // 20,070,400
  const size_t BUF2  = 20070400;    //  5,017,600
  const size_t IMRES = 25088000;    //  5,017,600 (persists)
  const size_t BUF3  = 30105600;    //  2,508,800
  const size_t BUF4  = 32614400;    //  1,254,400
  const size_t S     = 33868800;    // small arena
  float* t_c32   = ws + S;
  float* t_p32   = ws + S + 313600;
  float* t_c8    = ws + S + 387584;
  float* t_p8    = ws + S + 406080;
  float* t_c1    = ws + S + 410176;
  float* CAT     = ws + S + 410688;
  float* H1      = ws + S + 412240;
  float* H2      = ws + S + 413264;
  float* H3      = ws + S + 414288;
  float* ZTA     = ws + S + 415312;
  float* ZTB     = ws + S + 423504;
  float* XZ      = ws + S + 431696;
  float* XI      = ws + S + 464464;
  float* DBC     = ws + S + 480848;
  float* DELTA   = ws + S + 481616;
  float* Gb      = ws + S + 498000;
  float* CIN     = ws + S + 514384;
  float* C1b     = ws + S + 522580;
  float* C2b     = ws + S + 523604;
  float* DH      = ws + S + 524628;
  float* IMG     = ws + S + 532820;

  const int TB = 256;
  #define GRID1(n) dim3(cdiv_i((n), TB)), dim3(TB), 0, stream

  // ================= encoder =================
  conv3x3_gelu<<<GRID1(8 * 8 * 280 * 280)>>>(obs, e0w0, e0b0, ws + BUF2, 8, 1, 280, 280, 8);
  maxpool2_k  <<<GRID1(64 * 140 * 140)>>>(ws + BUF2, ws + BUF4, 8 * 8, 280, 280);
  run_conv_igemm(stream, ws + BUF4, e0w1, e0b1, ws + BUF1, 8, 8, 140, 140, 128);   // 8->128 WMMA
  maxpool2_k  <<<GRID1(1024 * 70 * 70)>>>(ws + BUF1, ws + IMRES, 8 * 128, 140, 140);
  run_conv_igemm(stream, ws + IMRES, e1w0, e1b0, ws + BUF3, 8, 128, 70, 70, 64);   // 128->64 WMMA
  maxpool2_k  <<<GRID1(512 * 35 * 35)>>>(ws + BUF3, ws + BUF4, 8 * 64, 70, 70);
  run_conv_igemm(stream, ws + BUF4, e1w1, e1b1, t_c32, 8, 64, 35, 35, 32);         // 64->32 WMMA
  maxpool2_k  <<<GRID1(256 * 17 * 17)>>>(t_c32, t_p32, 8 * 32, 35, 35);
  conv3x3_gelu<<<GRID1(8 * 8 * 17 * 17)>>>(t_p32, e1w2, e1b2, t_c8, 8, 32, 17, 17, 8);
  maxpool2_k  <<<GRID1(64 * 8 * 8)>>>(t_c8, t_p8, 8 * 8, 17, 17);
  conv3x3_gelu<<<GRID1(8 * 1 * 8 * 8)>>>(t_p8, e1w3, e1b3, t_c1, 8, 8, 8, 8, 1);

  // ================= lr_mlp =================
  copy_f32<<<GRID1(512)>>>(t_c1,    CAT,        512);
  copy_f32<<<GRID1(1024)>>>(state_t, CAT + 512,  1024);
  copy_f32<<<GRID1(10)>>>(action,   CAT + 1536, 10);
  run_gemm(stream, CAT, 1546, lw0, lb0, H1, 1024, 1, 1024, 1546, ACT_GELU);
  run_gemm(stream, H1, 1024,  lw1, lb1, H2, 1024, 1, 1024, 1024, ACT_GELU);
  run_gemm(stream, H2, 1024,  lw2, lb2, H3, 1024, 1, 1024, 1024, ACT_GELU);
  run_gemm(stream, H3, 1024,  lw3, lb3, ZTA, 8192, 1, 8192, 1024, ACT_NONE); // -> (8,1024)

  // ================= mamba x4 =================
  float* zt_in = ZTA;
  float* zt_out = ZTB;
  for (int bi = 0; bi < 4; ++bi) {
    int pb = 25 + 9 * bi;
    const float* in_proj  = IN(pb + 0);
    const float* conv_w   = IN(pb + 1);
    const float* conv_b   = IN(pb + 2);
    const float* x_proj   = IN(pb + 3);
    const float* dt_proj  = IN(pb + 4);
    const float* dt_bias  = IN(pb + 5);
    const float* A_log    = IN(pb + 6);
    const float* Dparam   = IN(pb + 7);
    const float* out_proj = IN(pb + 8);

    run_gemm(stream, zt_in, 1024, in_proj, nullptr, XZ, 4096, 8, 4096, 1024, ACT_NONE);
    mamba_conv_silu<<<GRID1(8 * 2048)>>>(XZ, conv_w, conv_b, XI, 8, 2048);
    run_gemm(stream, XI, 2048, x_proj, nullptr, DBC, 96, 8, 96, 2048, ACT_NONE);
    run_gemm(stream, DBC, 96, dt_proj, dt_bias, DELTA, 2048, 8, 2048, 64, ACT_SOFTPLUS);
    mamba_scan<<<dim3(cdiv_i(2048, TB)), dim3(TB), 0, stream>>>(
        DELTA, DBC, XI, XZ, A_log, Dparam, Gb, 8, 2048);
    run_gemm(stream, Gb, 2048, out_proj, nullptr, zt_out, 1024, 8, 1024, 2048, ACT_NONE);
    float* tmp = zt_in; zt_in = zt_out; zt_out = tmp;
  }
  float* ZT = zt_in;

  copy_f32<<<GRID1(1024)>>>(ZT + 7 * 1024, out + 627200, 1024);  // state_t_hat

  // ================= critics =================
  copy_f32<<<GRID1(8192)>>>(ZT, CIN, 8192);
  copy_f32<<<GRID1(1)>>>(steps_l, CIN + 8192, 1);
  copy_f32<<<GRID1(1)>>>(cur_r,   CIN + 8193, 1);
  run_gemm(stream, CIN, 8194, c0w0, c0b0, C1b, 1024, 1, 1024, 8194, ACT_GELU);
  run_gemm(stream, C1b, 1024, c0w1, c0b1, C2b, 1024, 1, 1024, 1024, ACT_GELU);
  run_gemm(stream, C2b, 1024, c0w2, c0b2, out + 628224, 1, 1, 1, 1024, ACT_NONE);
  run_gemm(stream, CIN, 8194, c1w0, c1b0, C1b, 1024, 1, 1024, 8194, ACT_GELU);
  run_gemm(stream, C1b, 1024, c1w1, c1b1, C2b, 1024, 1, 1024, 1024, ACT_GELU);
  run_gemm(stream, C2b, 1024, c1w2, c1b2, out + 628225, 1, 1, 1, 1024, ACT_NONE);

  // ================= decoder =================
  run_gemm(stream, ZT, 1024, dmw0, dmb0, DH, 1024, 8, 1024, 1024, ACT_GELU);
  run_gemm(stream, DH, 1024, dmw1, dmb1, IMG, 1225, 8, 1225, 1024, ACT_NONE); // (8,1,35,35)
  // dec0: 1->128 (K=4, too small for WMMA) direct, + image_res after gelu
  deconv4x4s2_gelu<<<GRID1(8 * 128 * 70 * 70)>>>(IMG, d0w, d0b, ws + IMRES,
                                                 ws + BUF2, 8, 1, 35, 35, 128);
  // dec1[0]: 128->64 via parity-split implicit-GEMM WMMA (K = 512)
  {
    dim3 g(cdiv_i(64, 128), cdiv_i(8 * 70 * 70, 16));
    for (int ph = 0; ph < 2; ++ph)
      for (int pw = 0; pw < 2; ++pw)
        deconv_igemm_wmma<<<g, dim3(256), 0, stream>>>(ws + BUF2, d1w, d1b, ws + BUF1,
                                                       8, 128, 70, 70, 64, ph, pw);
  }
  // dec1[1]: 64->1 (N=1) direct -> o
  deconv4x4s2_gelu<<<GRID1(8 * 1 * 280 * 280)>>>(ws + BUF1, d2w, d2b, nullptr,
                                                 out, 8, 64, 140, 140, 1);
  #undef GRID1
  #undef IN
}
